// ModifiedHunyuanVideoTransformerBlock_87840671138313
// MI455X (gfx1250) — matrix operand order repaired and checked by reading, not compiled, not run on hardware
//
#include <hip/hip_runtime.h>
#include <math.h>

// ---------------- CDNA5 types ----------------
typedef __bf16 bf16_t;
typedef __attribute__((ext_vector_type(16))) __bf16 v16bf;
typedef __attribute__((ext_vector_type(8)))  __bf16 v8bf;
typedef __attribute__((ext_vector_type(8)))  float  v8f;
typedef int i4v __attribute__((vector_size(16)));   // matches async-LDS builtin param

// ---------------- problem constants ----------------
#define DM     3072
#define NH     24
#define DH     128
#define FF     12288
#define S_IMG  2048
#define S_TXT  256
#define S_TOT  2304
#define SIXD   (6*DM)
#define LN_EPS 1e-6f

__device__ __forceinline__ bf16_t f2bf(float f) { return (bf16_t)f; }
__device__ __forceinline__ float  bf2f(bf16_t b) { return (float)b; }

// ---------------- optional CDNA5 async copy to LDS (ASYNCcnt path) ----
#if defined(__has_builtin)
#if __has_builtin(__builtin_amdgcn_global_load_async_to_lds_b128) && \
    __has_builtin(__builtin_amdgcn_s_wait_asynccnt)
#define USE_ASYNC_LDS 1
#endif
#endif
#ifndef USE_ASYNC_LDS
#define USE_ASYNC_LDS 0
#endif

#if USE_ASYNC_LDS
__device__ __forceinline__ void async_cp16(const bf16_t* g, bf16_t* l) {
    __builtin_amdgcn_global_load_async_to_lds_b128(
        (__attribute__((address_space(1))) i4v*)g,
        (__attribute__((address_space(3))) i4v*)l, 0, 0);
}
#define ASYNC_WAIT() __builtin_amdgcn_s_wait_asynccnt(0)
#else
#define ASYNC_WAIT()
#endif

// =====================================================================
// 1) AdaLN GEMV: out[n] = bias[n] + sum_k silu(temb[k]) * W[k][n]
// =====================================================================
__global__ __launch_bounds__(256) void ada_gemv_kernel(
    const float* __restrict__ temb, const float* __restrict__ W,
    const float* __restrict__ bias, float* __restrict__ out)
{
    __shared__ float st[DM];
    const int t = threadIdx.x;
    for (int j = t; j < DM; j += 256) {
        float v = temb[j];
        st[j] = v / (1.0f + __expf(-v));
    }
    __syncthreads();
    const int n = blockIdx.x * 256 + t;
    float acc = bias[n];
    #pragma unroll 4
    for (int k = 0; k < DM; ++k)
        acc += st[k] * W[(size_t)k * SIXD + n];
    out[n] = acc;
}

// =====================================================================
// 2) LayerNorm + AdaZero modulate:  out = LN(x)*(1+sc) + sh  (bf16 out)
// =====================================================================
__global__ __launch_bounds__(256) void ln_mod_kernel(
    const float* __restrict__ X, const float* __restrict__ sh,
    const float* __restrict__ sc, bf16_t* __restrict__ Out)
{
    const int row = blockIdx.x, t = threadIdx.x;
    const int lane = t & 31, w = t >> 5;
    const float* xr = X + (size_t)row * DM;
    float vals[12];
    float s1 = 0.f, s2 = 0.f;
    #pragma unroll
    for (int j = 0; j < 12; ++j) {
        float v = xr[t + j * 256];
        vals[j] = v; s1 += v; s2 += v * v;
    }
    #pragma unroll
    for (int off = 1; off < 32; off <<= 1) {
        s1 += __shfl_xor(s1, off, 32);
        s2 += __shfl_xor(s2, off, 32);
    }
    __shared__ float r1[8], r2[8];
    if (lane == 0) { r1[w] = s1; r2[w] = s2; }
    __syncthreads();
    float t1 = 0.f, t2 = 0.f;
    #pragma unroll
    for (int i = 0; i < 8; ++i) { t1 += r1[i]; t2 += r2[i]; }
    const float mu  = t1 * (1.0f / DM);
    const float var = t2 * (1.0f / DM) - mu * mu;
    const float inv = rsqrtf(var + LN_EPS);
    bf16_t* orow = Out + (size_t)row * DM;
    #pragma unroll
    for (int j = 0; j < 12; ++j) {
        int n = t + j * 256;
        float xn = (vals[j] - mu) * inv;
        orow[n] = f2bf(xn * (1.0f + sc[n]) + sh[n]);
    }
}

// =====================================================================
// 3) WMMA GEMM, double-buffered LDS, single barrier per K-step.
//    C[M,N] = A_bf16[M,K] @ W_f32[K,N] + bias, fused epilogue.
//    block = 256 thr (8 waves, 4x2), tile 128x128, K-step 32.
// =====================================================================
#define EP_BF16      0
#define EP_GELU_BF16 1
#define EP_RESID_F32 2

#define SA_STR 40   // 40 bf16 = 80B = 5*16B
#define SB_STR 40   // B transposed [n][k]

template <int EPI>
__global__ __launch_bounds__(256) void gemm_kernel(
    const bf16_t* __restrict__ A, const float* __restrict__ W,
    const float* __restrict__ bias, void* __restrict__ Cout,
    const float* __restrict__ resid, const float* __restrict__ gate,
    int M, int N, int K)
{
    __shared__ bf16_t sA[2][128 * SA_STR];
    __shared__ bf16_t sB[2][128 * SB_STR];

    const int tid = threadIdx.x;
    const int lane = tid & 31, w = tid >> 5;
    const int laneM = lane & 15, laneHi = lane >> 4;
    const int wm = w >> 1, wn = w & 1;            // 4 x 2 wave grid
    const int m0g = blockIdx.y * 128;
    const int n0g = blockIdx.x * 128;
    const int koffA = laneHi ? 8 : 0;
    const int koffB = laneHi ? 16 : 0;

    v8f acc[2][4];
    #pragma unroll
    for (int mi = 0; mi < 2; ++mi)
        #pragma unroll
        for (int ni = 0; ni < 4; ++ni)
            #pragma unroll
            for (int r = 0; r < 8; ++r) acc[mi][ni][r] = 0.0f;

    const int arow = tid >> 1, akp = (tid & 1) * 16;
    const bf16_t* agbase = A + (size_t)(m0g + arow) * K + akp;

    // ---- prologue: stage tile 0 into buffer 0 ----
    {
#if USE_ASYNC_LDS
        async_cp16(agbase,     &sA[0][arow * SA_STR + akp]);
        async_cp16(agbase + 8, &sA[0][arow * SA_STR + akp + 8]);
#else
        *(v8bf*)&sA[0][arow * SA_STR + akp]     = *(const v8bf*)agbase;
        *(v8bf*)&sA[0][arow * SA_STR + akp + 8] = *(const v8bf*)(agbase + 8);
#endif
        #pragma unroll
        for (int i = 0; i < 4; ++i) {
            int c = tid + i * 256;
            int kk = c >> 5, nn = (c & 31) << 2;
            const float4 wv = *(const float4*)(W + (size_t)kk * N + n0g + nn);
            sB[0][(nn + 0) * SB_STR + kk] = f2bf(wv.x);
            sB[0][(nn + 1) * SB_STR + kk] = f2bf(wv.y);
            sB[0][(nn + 2) * SB_STR + kk] = f2bf(wv.z);
            sB[0][(nn + 3) * SB_STR + kk] = f2bf(wv.w);
        }
        ASYNC_WAIT();
        __syncthreads();
    }

    const int nk = K / 32;
    for (int kt = 0; kt < nk; ++kt) {
        const int cur = kt & 1, nxt = cur ^ 1;
        const bool havenext = (kt + 1 < nk);
        const int k0n = (kt + 1) * 32;

        // ---- issue next tile: A via async-to-LDS (or regs), B via regs ----
        float4 wreg[4];
        v8bf areg0, areg1;
        if (havenext) {
            const bf16_t* ag = agbase + k0n;
#if USE_ASYNC_LDS
            async_cp16(ag,     &sA[nxt][arow * SA_STR + akp]);
            async_cp16(ag + 8, &sA[nxt][arow * SA_STR + akp + 8]);
#else
            areg0 = *(const v8bf*)ag;
            areg1 = *(const v8bf*)(ag + 8);
#endif
            #pragma unroll
            for (int i = 0; i < 4; ++i) {
                int c = tid + i * 256;
                int kk = c >> 5, nn = (c & 31) << 2;
                wreg[i] = *(const float4*)(W + (size_t)(k0n + kk) * N + n0g + nn);
            }
            if (kt + 2 < nk)  // -> global_prefetch_b8
                __builtin_prefetch(W + (size_t)(k0n + 32 + (tid >> 3)) * N + n0g + ((tid & 7) << 4), 0, 1);
        }

        // ---- fragments (contiguous k = 16*laneHi + s, A and B) + 8 WMMAs on current buffer ----
        v16bf af[2], bfr[4];
        #pragma unroll
        for (int mi = 0; mi < 2; ++mi) {
            const bf16_t* p = &sA[cur][(wm * 32 + mi * 16 + laneM) * SA_STR + koffB];
            v8bf lo = *(const v8bf*)p;
            v8bf hi = *(const v8bf*)(p + 8);
            #pragma unroll
            for (int i = 0; i < 8; ++i) { af[mi][i] = lo[i]; af[mi][i + 8] = hi[i]; }
        }
        #pragma unroll
        for (int ni = 0; ni < 4; ++ni)
            bfr[ni] = *(const v16bf*)&sB[cur][(wn * 64 + ni * 16 + laneM) * SB_STR + koffB];

        #pragma unroll
        for (int mi = 0; mi < 2; ++mi)
            #pragma unroll
            for (int ni = 0; ni < 4; ++ni)
                acc[mi][ni] = __builtin_amdgcn_wmma_f32_16x16x32_bf16(
                    false, af[mi], false, bfr[ni], (short)0, acc[mi][ni], false, false);

        // ---- land next tile into the other buffer ----
        if (havenext) {
#if !USE_ASYNC_LDS
            *(v8bf*)&sA[nxt][arow * SA_STR + akp]     = areg0;
            *(v8bf*)&sA[nxt][arow * SA_STR + akp + 8] = areg1;
#endif
            #pragma unroll
            for (int i = 0; i < 4; ++i) {
                int c = tid + i * 256;
                int kk = c >> 5, nn = (c & 31) << 2;
                sB[nxt][(nn + 0) * SB_STR + kk] = f2bf(wreg[i].x);
                sB[nxt][(nn + 1) * SB_STR + kk] = f2bf(wreg[i].y);
                sB[nxt][(nn + 2) * SB_STR + kk] = f2bf(wreg[i].z);
                sB[nxt][(nn + 3) * SB_STR + kk] = f2bf(wreg[i].w);
            }
        }
        ASYNC_WAIT();
        __syncthreads();
    }

    // ---- epilogue ----
    #pragma unroll
    for (int mi = 0; mi < 2; ++mi)
        #pragma unroll
        for (int ni = 0; ni < 4; ++ni) {
            const int gn = n0g + wn * 64 + ni * 16 + laneM;
            const float bv = bias[gn];
            #pragma unroll
            for (int r = 0; r < 8; ++r) {
                const int gm = m0g + wm * 32 + mi * 16 + laneHi * 8 + r;
                float v = acc[mi][ni][r] + bv;
                if (EPI == EP_GELU_BF16) {
                    float u = v;
                    v = 0.5f * u * (1.0f + tanhf(0.7978845608028654f * (u + 0.044715f * u * u * u)));
                    ((bf16_t*)Cout)[(size_t)gm * N + gn] = f2bf(v);
                } else if (EPI == EP_RESID_F32) {
                    ((float*)Cout)[(size_t)gm * N + gn] =
                        resid[(size_t)gm * N + gn] + gate[gn] * v;
                } else {
                    ((bf16_t*)Cout)[(size_t)gm * N + gn] = f2bf(v);
                }
            }
        }
}

// =====================================================================
// 4) RMS qk-norm + interleaved RoPE, in-place on bf16 [S][3072]
// =====================================================================
__global__ __launch_bounds__(128) void qknorm_rope_kernel(
    bf16_t* __restrict__ X, const float* __restrict__ nImg,
    const float* __restrict__ nTxt, const float* __restrict__ rc,
    const float* __restrict__ rs)
{
    const int s = blockIdx.x, h = blockIdx.y, t = threadIdx.x;
    const size_t idx = (size_t)s * DM + h * DH + t;
    const float v = bf2f(X[idx]);
    float ss = v * v;
    #pragma unroll
    for (int off = 1; off < 32; off <<= 1) ss += __shfl_xor(ss, off, 32);
    __shared__ float red[4];
    if ((t & 31) == 0) red[t >> 5] = ss;
    __syncthreads();
    const float tot = red[0] + red[1] + red[2] + red[3];
    const float inv = rsqrtf(tot * (1.0f / DH) + LN_EPS);
    const float* nw = (s < S_IMG) ? nImg : nTxt;
    float xn = v * inv * nw[t];
    float out = xn;
    if (s < S_IMG) {
        float part = __shfl_xor(xn, 1, 32);
        float c = rc[s * DH + t], sn = rs[s * DH + t];
        out = (t & 1) ? (xn * c + part * sn) : (xn * c - part * sn);
    }
    X[idx] = f2bf(out);
}

// =====================================================================
// 5) Flash attention, bf16 WMMA.  grid (S_TOT/128, NH), 256 thr (8 waves)
// =====================================================================
#define SK_STR 136  // 128 + 8 pad ; 272B = 17*16B
#define SV_STR 40

__global__ __launch_bounds__(256) void attn_kernel(
    const bf16_t* __restrict__ Q, const bf16_t* __restrict__ Kb,
    const bf16_t* __restrict__ Vb, bf16_t* __restrict__ Ob)
{
    __shared__ bf16_t sK[32 * SK_STR];
    __shared__ bf16_t sVt[128 * SV_STR];     // V transposed [dh][kv]
    __shared__ bf16_t sP[8 * 16 * SV_STR];   // per-wave P staging

    const int tid = threadIdx.x;
    const int lane = tid & 31, w = tid >> 5;
    const int laneM = lane & 15, laneHi = lane >> 4;
    const int h = blockIdx.y;
    const int q0 = blockIdx.x * 128 + w * 16;
    const int koffA = laneHi ? 8 : 0;
    const int koffB = laneHi ? 16 : 0;
    const float SC = 0.08838834764831845f;   // 1/sqrt(128)

    v16bf qf[4];
    {
        const bf16_t* qp = Q + (size_t)(q0 + laneM) * DM + h * DH;
        #pragma unroll
        for (int kc = 0; kc < 4; ++kc) {
            v8bf lo = *(const v8bf*)(qp + kc * 32 + koffB);
            v8bf hi = *(const v8bf*)(qp + kc * 32 + 8 + koffB);
            #pragma unroll
            for (int i = 0; i < 8; ++i) { qf[kc][i] = lo[i]; qf[kc][i + 8] = hi[i]; }
        }
    }

    v8f o[8];
    #pragma unroll
    for (int ni = 0; ni < 8; ++ni)
        #pragma unroll
        for (int r = 0; r < 8; ++r) o[ni][r] = 0.0f;
    float mRow[8], lRow[8];
    #pragma unroll
    for (int r = 0; r < 8; ++r) { mRow[r] = -1e30f; lRow[r] = 0.0f; }

    const int ldRow = tid >> 3, ldPart = (tid & 7) << 4;

    for (int j = 0; j < S_TOT; j += 32) {
        // ---- stage K (async direct copy) and V (transposed, VALU) ----
        {
            const bf16_t* kg = Kb + (size_t)(j + ldRow) * DM + h * DH + ldPart;
            bf16_t* lk = &sK[ldRow * SK_STR + ldPart];
#if USE_ASYNC_LDS
            async_cp16(kg, lk);
            async_cp16(kg + 8, lk + 8);
#else
            *(v8bf*)lk       = *(const v8bf*)kg;
            *(v8bf*)(lk + 8) = *(const v8bf*)(kg + 8);
#endif
            const bf16_t* vg = Vb + (size_t)(j + ldRow) * DM + h * DH + ldPart;
            v8bf v0 = *(const v8bf*)vg, v1 = *(const v8bf*)(vg + 8);
            #pragma unroll
            for (int i = 0; i < 8; ++i) {
                sVt[(ldPart + i) * SV_STR + ldRow]     = v0[i];
                sVt[(ldPart + 8 + i) * SV_STR + ldRow] = v1[i];
            }
        }
        ASYNC_WAIT();
        __syncthreads();

        // ---- S = Q K^T ----
        v8f s0, s1;
        #pragma unroll
        for (int r = 0; r < 8; ++r) { s0[r] = 0.0f; s1[r] = 0.0f; }
        #pragma unroll
        for (int kc = 0; kc < 4; ++kc) {
            v16bf b0 = *(const v16bf*)&sK[laneM * SK_STR + kc * 32 + koffB];
            v16bf b1 = *(const v16bf*)&sK[(16 + laneM) * SK_STR + kc * 32 + koffB];
            s0 = __builtin_amdgcn_wmma_f32_16x16x32_bf16(false, qf[kc], false, b0, (short)0, s0, false, false);
            s1 = __builtin_amdgcn_wmma_f32_16x16x32_bf16(false, qf[kc], false, b1, (short)0, s1, false, false);
        }

        // ---- online softmax (row groups live in 16-lane halves) ----
        float rm[8];
        #pragma unroll
        for (int r = 0; r < 8; ++r) {
            s0[r] *= SC; s1[r] *= SC;
            rm[r] = fmaxf(s0[r], s1[r]);
        }
        #pragma unroll
        for (int off = 1; off < 16; off <<= 1)
            #pragma unroll
            for (int r = 0; r < 8; ++r)
                rm[r] = fmaxf(rm[r], __shfl_xor(rm[r], off, 32));
        float alpha[8], rsum[8];
        #pragma unroll
        for (int r = 0; r < 8; ++r) {
            float mn = fmaxf(mRow[r], rm[r]);
            alpha[r] = __expf(mRow[r] - mn);
            mRow[r] = mn;
            float p0 = __expf(s0[r] - mn), p1 = __expf(s1[r] - mn);
            s0[r] = p0; s1[r] = p1;
            rsum[r] = p0 + p1;
        }
        #pragma unroll
        for (int off = 1; off < 16; off <<= 1)
            #pragma unroll
            for (int r = 0; r < 8; ++r)
                rsum[r] += __shfl_xor(rsum[r], off, 32);
        #pragma unroll
        for (int r = 0; r < 8; ++r) lRow[r] = lRow[r] * alpha[r] + rsum[r];

        // ---- P: C-layout -> A-layout via per-wave LDS staging ----
        bf16_t* pp = &sP[w * 16 * SV_STR];
        #pragma unroll
        for (int r = 0; r < 8; ++r) {
            int m = laneHi * 8 + r;
            pp[m * SV_STR + laneM]      = f2bf(s0[r]);
            pp[m * SV_STR + 16 + laneM] = f2bf(s1[r]);
        }
        #pragma unroll
        for (int ni = 0; ni < 8; ++ni)
            #pragma unroll
            for (int r = 0; r < 8; ++r) o[ni][r] *= alpha[r];

        v16bf pa;
        {
            const bf16_t* p = &pp[laneM * SV_STR + koffB];
            v8bf lo = *(const v8bf*)p, hi = *(const v8bf*)(p + 8);
            #pragma unroll
            for (int i = 0; i < 8; ++i) { pa[i] = lo[i]; pa[i + 8] = hi[i]; }
        }

        // ---- O += P V ----
        #pragma unroll
        for (int ni = 0; ni < 8; ++ni) {
            v16bf bv = *(const v16bf*)&sVt[(ni * 16 + laneM) * SV_STR + koffB];
            o[ni] = __builtin_amdgcn_wmma_f32_16x16x32_bf16(false, pa, false, bv, (short)0, o[ni], false, false);
        }
        __syncthreads();
    }

    // ---- normalize + write [s][3072] bf16 ----
    #pragma unroll
    for (int ni = 0; ni < 8; ++ni)
        #pragma unroll
        for (int r = 0; r < 8; ++r) {
            int row = q0 + laneHi * 8 + r;
            Ob[(size_t)row * DM + h * DH + ni * 16 + laneM] = f2bf(o[ni][r] / lRow[r]);
        }
}

// =====================================================================
// host: launch sequence
// =====================================================================
extern "C" void kernel_launch(void* const* d_in, const int* in_sizes, int n_in,
                              void* d_out, int out_size, void* d_ws, size_t ws_size,
                              hipStream_t stream)
{
    const float* hidden = (const float*)d_in[0];
    const float* enc    = (const float*)d_in[1];
    const float* temb   = (const float*)d_in[2];
    const float* rcos   = (const float*)d_in[3];
    const float* rsin   = (const float*)d_in[4];
    const float* ada_w  = (const float*)d_in[5];
    const float* ada_b  = (const float*)d_in[6];
    const float* adac_w = (const float*)d_in[7];
    const float* adac_b = (const float*)d_in[8];
    const float* wq = (const float*)d_in[9],  *bq = (const float*)d_in[10];
    const float* wk = (const float*)d_in[11], *bk = (const float*)d_in[12];
    const float* wv = (const float*)d_in[13], *bv = (const float*)d_in[14];
    const float* qn = (const float*)d_in[15], *kn = (const float*)d_in[16];
    const float* awq = (const float*)d_in[17], *abq = (const float*)d_in[18];
    const float* awk = (const float*)d_in[19], *abk = (const float*)d_in[20];
    const float* awv = (const float*)d_in[21], *abv = (const float*)d_in[22];
    const float* aqn = (const float*)d_in[23], *akn = (const float*)d_in[24];
    const float* wo  = (const float*)d_in[25], *bo  = (const float*)d_in[26];
    const float* awo = (const float*)d_in[27], *abo = (const float*)d_in[28];
    const float* fw1 = (const float*)d_in[29], *fb1 = (const float*)d_in[30];
    const float* fw2 = (const float*)d_in[31], *fb2 = (const float*)d_in[32];
    const float* cw1 = (const float*)d_in[33], *cb1 = (const float*)d_in[34];
    const float* cw2 = (const float*)d_in[35], *cb2 = (const float*)d_in[36];

    // ---- workspace layout (bytes) ----
    char* ws = (char*)d_ws;
    float*  eimg = (float*)(ws + 0);                         // 6D f32
    float*  etxt = (float*)(ws + 73728);
    bf16_t* abuf = (bf16_t*)(ws + 147456);                   // [2304][3072] bf16
    const size_t ACT = (size_t)S_TOT * DM * sizeof(bf16_t);  // 14,155,776
    char* qkvo = ws + 147456 + ACT;
    bf16_t* qbuf = (bf16_t*)(qkvo + 0 * ACT);
    bf16_t* kbuf = (bf16_t*)(qkvo + 1 * ACT);
    bf16_t* vbuf = (bf16_t*)(qkvo + 2 * ACT);
    bf16_t* obuf = (bf16_t*)(qkvo + 3 * ACT);
    bf16_t* hbuf = (bf16_t*)qkvo;                            // aliases q/k/v/o (dead by then)

    float* out_f   = (float*)d_out;                          // h rows [0,2048), e rows [2048,2304)
    bf16_t* abuf_t = abuf + (size_t)S_IMG * DM;
    bf16_t* qbuf_t = qbuf + (size_t)S_IMG * DM;
    bf16_t* kbuf_t = kbuf + (size_t)S_IMG * DM;
    bf16_t* vbuf_t = vbuf + (size_t)S_IMG * DM;
    bf16_t* obuf_t = obuf + (size_t)S_IMG * DM;
    bf16_t* hbuf_t = hbuf + (size_t)S_IMG * FF;

    const dim3 B256(256), B128(128);
    const dim3 gI(DM / 128, S_IMG / 128);   // (24,16)
    const dim3 gT(DM / 128, S_TXT / 128);   // (24,2)
    const dim3 gFI(FF / 128, S_IMG / 128);  // (96,16)
    const dim3 gFT(FF / 128, S_TXT / 128);  // (96,2)

    // 1) AdaLN vectors (6D each): [sh_msa, sc_msa, g_msa, sh_mlp, sc_mlp, g_mlp]
    ada_gemv_kernel<<<SIXD / 256, B256, 0, stream>>>(temb, ada_w, ada_b, eimg);
    ada_gemv_kernel<<<SIXD / 256, B256, 0, stream>>>(temb, adac_w, adac_b, etxt);

    // 2) pre-attn LN+mod -> bf16 activations
    ln_mod_kernel<<<S_IMG, B256, 0, stream>>>(hidden, eimg + 0 * DM, eimg + 1 * DM, abuf);
    ln_mod_kernel<<<S_TXT, B256, 0, stream>>>(enc,    etxt + 0 * DM, etxt + 1 * DM, abuf_t);

    // 3) QKV projections (bf16 out)
    gemm_kernel<EP_BF16><<<gI, B256, 0, stream>>>(abuf,   wq,  bq,  qbuf,   nullptr, nullptr, S_IMG, DM, DM);
    gemm_kernel<EP_BF16><<<gT, B256, 0, stream>>>(abuf_t, awq, abq, qbuf_t, nullptr, nullptr, S_TXT, DM, DM);
    gemm_kernel<EP_BF16><<<gI, B256, 0, stream>>>(abuf,   wk,  bk,  kbuf,   nullptr, nullptr, S_IMG, DM, DM);
    gemm_kernel<EP_BF16><<<gT, B256, 0, stream>>>(abuf_t, awk, abk, kbuf_t, nullptr, nullptr, S_TXT, DM, DM);
    gemm_kernel<EP_BF16><<<gI, B256, 0, stream>>>(abuf,   wv,  bv,  vbuf,   nullptr, nullptr, S_IMG, DM, DM);
    gemm_kernel<EP_BF16><<<gT, B256, 0, stream>>>(abuf_t, awv, abv, vbuf_t, nullptr, nullptr, S_TXT, DM, DM);

    // 4) RMS qk-norm (+RoPE on image tokens), in place
    qknorm_rope_kernel<<<dim3(S_TOT, NH), B128, 0, stream>>>(qbuf, qn, aqn, rcos, rsin);
    qknorm_rope_kernel<<<dim3(S_TOT, NH), B128, 0, stream>>>(kbuf, kn, akn, rcos, rsin);

    // 5) joint flash attention
    attn_kernel<<<dim3(S_TOT / 128, NH), B256, 0, stream>>>(qbuf, kbuf, vbuf, obuf);

    // 6) out-proj + gated residual -> d_out (f32)
    gemm_kernel<EP_RESID_F32><<<gI, B256, 0, stream>>>(obuf,   wo,  bo,  out_f,                      hidden, eimg + 2 * DM, S_IMG, DM, DM);
    gemm_kernel<EP_RESID_F32><<<gT, B256, 0, stream>>>(obuf_t, awo, abo, out_f + (size_t)S_IMG * DM, enc,    etxt + 2 * DM, S_TXT, DM, DM);

    // 7) pre-MLP LN+mod
    ln_mod_kernel<<<S_IMG, B256, 0, stream>>>(out_f,                      eimg + 3 * DM, eimg + 4 * DM, abuf);
    ln_mod_kernel<<<S_TXT, B256, 0, stream>>>(out_f + (size_t)S_IMG * DM, etxt + 3 * DM, etxt + 4 * DM, abuf_t);

    // 8) MLP fc1 + tanh-GELU (bf16)
    gemm_kernel<EP_GELU_BF16><<<gFI, B256, 0, stream>>>(abuf,   fw1, fb1, hbuf,   nullptr, nullptr, S_IMG, FF, DM);
    gemm_kernel<EP_GELU_BF16><<<gFT, B256, 0, stream>>>(abuf_t, cw1, cb1, hbuf_t, nullptr, nullptr, S_TXT, FF, DM);

    // 9) MLP fc2 + gated residual, final write to d_out
    gemm_kernel<EP_RESID_F32><<<gI, B256, 0, stream>>>(hbuf,   fw2, fb2, out_f,                      out_f,                      eimg + 5 * DM, S_IMG, DM, FF);
    gemm_kernel<EP_RESID_F32><<<gT, B256, 0, stream>>>(hbuf_t, cw2, cb2, out_f + (size_t)S_IMG * DM, out_f + (size_t)S_IMG * DM, etxt + 5 * DM, S_TXT, DM, FF);

    (void)in_sizes; (void)n_in; (void)out_size; (void)ws_size;
}
